// LLMLayer_10187662426436
// MI455X (gfx1250) — matrix operand order repaired and checked by reading, not compiled, not run on hardware
//
#include <hip/hip_runtime.h>
#include <hip/hip_bf16.h>

// ---------------------------------------------------------------------------
// LLM transformer layer for gfx1250 (MI455X).
// All heavy GEMMs: v_wmma_f32_16x16x32_bf16, bf16 operands staged to LDS via
// GLOBAL_LOAD_ASYNC_TO_LDS_B128 (when available) with double buffering.
// Weights are transposed+converted to bf16 K-major once per call, so every
// GEMM is "NT": both A and B fragments are contiguous-K ds_load_b128 pairs.
// ---------------------------------------------------------------------------

#define S_LEN  2048
#define HID    2048
#define HD     128
#define QH     16
#define KVH    4
#define QKV_N  3072
#define FF     8192
#define LORA_SCALE 2.0f
#define SCALER 0.08838834764831845f  // 1/sqrt(128)

#define AS1 __attribute__((address_space(1)))
#define AS3 __attribute__((address_space(3)))

#if defined(__has_builtin)
#  if __has_builtin(__builtin_amdgcn_global_load_async_to_lds_b128)
#    define HAS_ASYNC 1
#  endif
#endif
#ifndef HAS_ASYNC
#  define HAS_ASYNC 0
#endif

typedef __attribute__((ext_vector_type(16))) __bf16 bf16x16;
typedef __attribute__((ext_vector_type(8)))  float  v8f;
typedef int v4i __attribute__((vector_size(16)));   // matches builtin param type

union FragBF { uint4 q[2]; bf16x16 v; };

__device__ inline unsigned short f2bf(float f) {
  unsigned u = __float_as_uint(f);
  unsigned r = u + 0x7FFFu + ((u >> 16) & 1u);   // round-to-nearest-even
  return (unsigned short)(r >> 16);
}
__device__ inline float b2f(unsigned short b) {
  return __uint_as_float((unsigned)b << 16);
}

__device__ inline void wait_async0() {
#if defined(__has_builtin)
#  if __has_builtin(__builtin_amdgcn_s_wait_asynccnt)
  __builtin_amdgcn_s_wait_asynccnt(0);
#  else
  asm volatile("s_wait_asynccnt 0" ::: "memory");
#  endif
#else
  asm volatile("s_wait_asynccnt 0" ::: "memory");
#endif
}

// A fragment (16x32 bf16): lane<16 -> row m, K{0..7,16..23}; lane>=16 -> K{8..15,24..31}
__device__ inline bf16x16 frag_a(const unsigned short* sm, int row0, int ld) {
  int lane = threadIdx.x & 31;
  const unsigned short* p = sm + (row0 + (lane & 15)) * ld + ((lane < 16) ? 0 : 8);
  FragBF f;
  f.q[0] = *(const uint4*)(p);
  f.q[1] = *(const uint4*)(p + 16);
  return f.v;
}
// B fragment (32x16 bf16), Bs[col][k]: lane<16 -> col n, K{0..7,16..23}; lane>=16 -> K{8..15,24..31}
__device__ inline bf16x16 frag_b(const unsigned short* sm, int col0, int ld) {
  int lane = threadIdx.x & 31;
  const unsigned short* p = sm + (col0 + (lane & 15)) * ld + ((lane < 16) ? 0 : 8);
  FragBF f;
  f.q[0] = *(const uint4*)(p);
  f.q[1] = *(const uint4*)(p + 16);
  return f.v;
}

#define BM 128
#define BN 128
#define BK 32
#define TLD 40                  // BK + 8 pad ushorts; 80B row stride keeps 16B align
#define TILE_USH (BM * TLD)

// C[M,N] = alpha*(A x B^T) + bias + LORA_SCALE*(loraT@loraB + loraBB) + mask + resid
// A: bf16 [M,K] row-major. B: bf16 [N,K] row-major (K-major / "NT").
__global__ void __launch_bounds__(256)
k_gemm_nt(const unsigned short* __restrict__ A, int lda,
          const unsigned short* __restrict__ B, int ldb,
          float* __restrict__ C, int ldc,
          int M, int N, int K,
          const float* __restrict__ bias,
          const float* __restrict__ loraT,   // [M,16] fp32
          const float* __restrict__ loraB,   // [16,N] fp32 (row stride N)
          const float* __restrict__ loraBB,  // [N]
          const float* __restrict__ resid,   // [M,ldc] fp32
          float alpha, int causal)
{
  __shared__ __align__(16) unsigned short As[2][TILE_USH];
  __shared__ __align__(16) unsigned short Bs[2][TILE_USH];

  const int t = threadIdx.x, lane = t & 31, wave = t >> 5;
  const int wm = wave >> 2, wn = wave & 3;            // 2 x 4 waves
  const int rowBase = blockIdx.y * BM, colBase = blockIdx.x * BN;

  v8f acc[4][2] = {};
  const int nk = K / BK;

  // staging: 128 rows x 32 bf16 (64B/row = 4 x 16B); 512 tasks, 2 per thread
  auto stage = [&](int buf, int k0) {
#pragma unroll
    for (int it = 0; it < 2; ++it) {
      int idx = t + it * 256;
      int r = idx >> 2, q = idx & 3;
      const unsigned short* ga = A + (long)(rowBase + r) * lda + k0 + q * 8;
      const unsigned short* gb = B + (long)(colBase + r) * ldb + k0 + q * 8;
      unsigned short* la = &As[buf][r * TLD + q * 8];
      unsigned short* lb = &Bs[buf][r * TLD + q * 8];
#if HAS_ASYNC
      __builtin_amdgcn_global_load_async_to_lds_b128((AS1 v4i*)ga, (AS3 v4i*)la, 0, 0);
      __builtin_amdgcn_global_load_async_to_lds_b128((AS1 v4i*)gb, (AS3 v4i*)lb, 0, 0);
#else
      *(uint4*)la = *(const uint4*)ga;
      *(uint4*)lb = *(const uint4*)gb;
#endif
    }
  };

  stage(0, 0);
  for (int kc = 0; kc < nk; ++kc) {
    const int cur = kc & 1;
#if HAS_ASYNC
    wait_async0();
#endif
    __syncthreads();
    if (kc + 1 < nk) stage(cur ^ 1, (kc + 1) * BK);

    const unsigned short* as = As[cur];
    const unsigned short* bs = Bs[cur];
    bf16x16 bfr[2];
#pragma unroll
    for (int j = 0; j < 2; ++j) bfr[j] = frag_b(bs, wn * 32 + j * 16, TLD);
#pragma unroll
    for (int i = 0; i < 4; ++i) {
      bf16x16 afr = frag_a(as, wm * 64 + i * 16, TLD);
#pragma unroll
      for (int j = 0; j < 2; ++j)
        acc[i][j] = __builtin_amdgcn_wmma_f32_16x16x32_bf16(
            false, afr, false, bfr[j], (short)0, acc[i][j], false, false);
    }
    __syncthreads();
  }

  // epilogue: D layout VGPR v -> (m = v + (lane>=16?8:0), n = lane&15)
  const int lhalf = (lane >= 16) ? 8 : 0;
#pragma unroll
  for (int j = 0; j < 2; ++j) {
    int colg = colBase + wn * 32 + j * 16 + (lane & 15);
    float lb16[16], bb = 0.f, bi = 0.f;
    if (loraT) {
#pragma unroll
      for (int r = 0; r < 16; ++r) lb16[r] = loraB[(long)r * N + colg];
      bb = loraBB[colg];
    }
    if (bias) bi = bias[colg];
#pragma unroll
    for (int i = 0; i < 4; ++i) {
      int row0 = rowBase + wm * 64 + i * 16 + lhalf;
#pragma unroll
      for (int v = 0; v < 8; ++v) {
        int row = row0 + v;
        float val = acc[i][j][v] * alpha + bi;
        if (loraT) {
          const float* tr = loraT + (long)row * 16;
          float d = bb;
#pragma unroll
          for (int r = 0; r < 16; ++r) d += tr[r] * lb16[r];
          val += LORA_SCALE * d;
        }
        if (causal && colg > row) val += -1e9f;
        if (resid) val += resid[(long)row * ldc + colg];
        C[(long)row * ldc + colg] = val;
      }
    }
  }
}

// ---------------------------------------------------------------------------
// transpose + fp32->bf16:  Wt[n*K + k] = bf16(W[k*ldw + n]),  K x N tile 32x32
__global__ void __launch_bounds__(256)
k_w2bf16t(const float* __restrict__ W, unsigned short* __restrict__ Wt,
          int K, int N, int ldw)
{
  __shared__ unsigned short ls[32][33];
  int n0 = blockIdx.x * 32, k0 = blockIdx.y * 32;
  int t = threadIdx.x;
#pragma unroll
  for (int p = 0; p < 4; ++p) {
    int e = t + p * 256;
    int kk = e >> 5, nn = e & 31;
    ls[kk][nn] = f2bf(W[(long)(k0 + kk) * ldw + n0 + nn]);
  }
  __syncthreads();
#pragma unroll
  for (int p = 0; p < 4; ++p) {
    int e = t + p * 256;
    int nn = e >> 5, kk = e & 31;
    Wt[(long)(n0 + nn) * K + k0 + kk] = ls[kk][nn];
  }
}

__global__ void __launch_bounds__(256)
k_rmsnorm_bf16(const float* __restrict__ x, const float* __restrict__ w,
               unsigned short* __restrict__ o, int H)
{
  __shared__ float red[256];
  long row = blockIdx.x;
  int t = threadIdx.x;
  float s = 0.f;
  for (int c = t; c < H; c += 256) { float v = x[row * H + c]; s += v * v; }
  red[t] = s; __syncthreads();
  for (int st = 128; st > 0; st >>= 1) {
    if (t < st) red[t] += red[t + st];
    __syncthreads();
  }
  float inv = rsqrtf(red[0] / (float)H + 1e-6f);
  for (int c = t; c < H; c += 256) o[row * H + c] = f2bf(x[row * H + c] * inv * w[c]);
}

// out[row,0..15] = act( bf16 X[row,:K] @ W[K,16] + b )
__global__ void __launch_bounds__(256)
k_smallproj(const unsigned short* __restrict__ X, const float* __restrict__ W,
            const float* __restrict__ b, float* __restrict__ out, int K, int act)
{
  __shared__ float xs[FF];
  __shared__ float part[16][16];
  long row = blockIdx.x;
  int t = threadIdx.x;
  for (int k = t; k < K; k += 256) xs[k] = b2f(X[row * K + k]);
  __syncthreads();
  int n = t & 15, p = t >> 4;
  float acc = 0.f;
  for (int k = p; k < K; k += 16) acc += xs[k] * W[k * 16 + n];
  part[p][n] = acc;
  __syncthreads();
  if (t < 16) {
    float s = b[t];
    for (int pp = 0; pp < 16; ++pp) s += part[pp][t];
    if (act) s = 1.f / (1.f + __expf(-s));
    out[row * 16 + t] = s;
  }
}

// interleaved-pair RoPE; writes bf16 q/k
__global__ void k_rope(const float* __restrict__ qkv,
                       unsigned short* __restrict__ qr, unsigned short* __restrict__ kr)
{
  int s = blockIdx.x, h = blockIdx.y, t = threadIdx.x;   // t: 0..63 pairs
  float freq = __powf(10000.f, -(float)t / 64.f);
  float sn, cs;
  __sincosf(freq * (float)s, &sn, &cs);
  const float* src; unsigned short* dst;
  if (h < QH) {
    src = qkv + (long)s * QKV_N + h * HD;
    dst = qr + (long)s * (QH * HD) + h * HD;
  } else {
    src = qkv + (long)s * QKV_N + QH * HD + (h - QH) * HD;
    dst = kr + (long)s * (KVH * HD) + (h - QH) * HD;
  }
  float x0 = src[2 * t], x1 = src[2 * t + 1];
  dst[2 * t]     = f2bf(x0 * cs - x1 * sn);
  dst[2 * t + 1] = f2bf(x1 * cs + x0 * sn);
}

// fp32 scores row -> bf16 probs row
__global__ void __launch_bounds__(256)
k_softmax_bf16(const float* __restrict__ sc, unsigned short* __restrict__ pb, int N)
{
  __shared__ float red[256];
  long row = blockIdx.x;
  const float* p = sc + row * N;
  int t = threadIdx.x;
  float m = -3.4e38f;
  for (int c = t; c < N; c += 256) m = fmaxf(m, p[c]);
  red[t] = m; __syncthreads();
  for (int st = 128; st > 0; st >>= 1) {
    if (t < st) red[t] = fmaxf(red[t], red[t + st]);
    __syncthreads();
  }
  m = red[0]; __syncthreads();
  float s = 0.f;
  for (int c = t; c < N; c += 256) s += __expf(p[c] - m);
  red[t] = s; __syncthreads();
  for (int st = 128; st > 0; st >>= 1) {
    if (t < st) red[t] += red[t + st];
    __syncthreads();
  }
  float inv = 1.f / red[0];
  for (int c = t; c < N; c += 256) pb[row * N + c] = f2bf(__expf(p[c] - m) * inv);
}

__global__ void k_gateapply(const float* __restrict__ av, const float* __restrict__ g,
                            unsigned short* __restrict__ ob)
{
  long i = (long)blockIdx.x * blockDim.x + threadIdx.x;   // S*HID
  long s = i / HID;
  int  h = (int)((i % HID) / HD);
  ob[i] = f2bf(av[i] * g[s * QH + h]);
}

__global__ void k_geluup(const float* __restrict__ gu, unsigned short* __restrict__ hff)
{
  long i = (long)blockIdx.x * blockDim.x + threadIdx.x;   // S*FF
  long s = i / FF, j = i % FF;
  float g = gu[s * (2 * FF) + j];
  float u = gu[s * (2 * FF) + FF + j];
  float t = 0.7978845608028654f * (g + 0.044715f * g * g * g);
  hff[i] = f2bf(0.5f * g * (1.f + tanhf(t)) * u);
}

// ---------------------------------------------------------------------------
extern "C" void kernel_launch(void* const* d_in, const int* in_sizes, int n_in,
                              void* d_out, int out_size, void* d_ws, size_t ws_size,
                              hipStream_t stream)
{
  const float* x      = (const float*)d_in[0];
  const float* n1w    = (const float*)d_in[2];
  const float* n2w    = (const float*)d_in[3];
  const float* qkv_w  = (const float*)d_in[4];
  const float* qkv_b  = (const float*)d_in[5];
  const float* lqa_w  = (const float*)d_in[6];
  const float* lqa_b  = (const float*)d_in[7];
  const float* lqb_w  = (const float*)d_in[8];
  const float* lqb_b  = (const float*)d_in[9];
  const float* o_w    = (const float*)d_in[10];
  const float* o_b    = (const float*)d_in[11];
  const float* loa_w  = (const float*)d_in[12];
  const float* loa_b  = (const float*)d_in[13];
  const float* lob_w  = (const float*)d_in[14];
  const float* lob_b  = (const float*)d_in[15];
  const float* gate_w = (const float*)d_in[16];
  const float* gate_b = (const float*)d_in[17];
  const float* gu_w   = (const float*)d_in[18];
  const float* gu_b   = (const float*)d_in[19];
  const float* lga_w  = (const float*)d_in[20];
  const float* lga_b  = (const float*)d_in[21];
  const float* lgb_w  = (const float*)d_in[22];
  const float* lgb_b  = (const float*)d_in[23];
  const float* dn_w   = (const float*)d_in[24];
  const float* dn_b   = (const float*)d_in[25];
  const float* ldna_w = (const float*)d_in[26];
  const float* ldna_b = (const float*)d_in[27];
  const float* ldnb_w = (const float*)d_in[28];
  const float* ldnb_b = (const float*)d_in[29];
  float* out = (float*)d_out;

  // ---- workspace carve-up ----
  char* wp = (char*)d_ws;
  auto alloc = [&](size_t bytes) {
    void* r = (void*)wp;
    wp += (bytes + 255) & ~(size_t)255;
    return r;
  };
  const long SL = S_LEN;
  float* qkv   = (float*)alloc(SL * QKV_N * 4);
  float* sc    = (float*)alloc(SL * SL * 4);
  float* av    = (float*)alloc(SL * HID * 4);
  float* hres  = (float*)alloc(SL * HID * 4);
  float* gu    = (float*)alloc(SL * 2 * FF * 4);
  float* gates = (float*)alloc(SL * 16 * 4);
  float* tq    = (float*)alloc(SL * 16 * 4);
  float* to_   = (float*)alloc(SL * 16 * 4);
  float* tg    = (float*)alloc(SL * 16 * 4);
  float* td    = (float*)alloc(SL * 16 * 4);
  unsigned short* xn     = (unsigned short*)alloc(SL * HID * 2);
  unsigned short* xn2    = (unsigned short*)alloc(SL * HID * 2);
  unsigned short* qr     = (unsigned short*)alloc(SL * QH * HD * 2);
  unsigned short* kr     = (unsigned short*)alloc(SL * KVH * HD * 2);
  unsigned short* vt     = (unsigned short*)alloc((long)KVH * HD * SL * 2);
  unsigned short* probs  = (unsigned short*)alloc(SL * SL * 2);
  unsigned short* av_bf  = (unsigned short*)alloc(SL * HID * 2);
  unsigned short* hff    = (unsigned short*)alloc(SL * FF * 2);
  unsigned short* qkv_wt = (unsigned short*)alloc((long)QKV_N * HID * 2);
  unsigned short* o_wt   = (unsigned short*)alloc((long)HID * HID * 2);
  unsigned short* gu_wt  = (unsigned short*)alloc((long)2 * FF * HID * 2);
  unsigned short* dn_wt  = (unsigned short*)alloc((long)HID * FF * 2);

  dim3 blk(256);
  auto gemm = [&](const unsigned short* A, int lda, const unsigned short* B, int ldb,
                  float* C, int ldc, int M, int N, int K,
                  const float* bias, const float* lT, const float* lB,
                  const float* lBB, const float* res, float alpha, int causal) {
    dim3 g(N / BN, M / BM);
    k_gemm_nt<<<g, blk, 0, stream>>>(A, lda, B, ldb, C, ldc, M, N, K,
                                     bias, lT, lB, lBB, res, alpha, causal);
  };

  // ---- one-time weight transpose + bf16 convert ----
  k_w2bf16t<<<dim3(QKV_N / 32, HID / 32), blk, 0, stream>>>(qkv_w, qkv_wt, HID, QKV_N, QKV_N);
  k_w2bf16t<<<dim3(HID / 32, HID / 32),   blk, 0, stream>>>(o_w,  o_wt,  HID, HID, HID);
  k_w2bf16t<<<dim3(2 * FF / 32, HID / 32),blk, 0, stream>>>(gu_w, gu_wt, HID, 2 * FF, 2 * FF);
  k_w2bf16t<<<dim3(HID / 32, FF / 32),    blk, 0, stream>>>(dn_w, dn_wt, FF, HID, HID);

  // ---- attention branch ----
  k_rmsnorm_bf16<<<S_LEN, blk, 0, stream>>>(x, n1w, xn, HID);
  k_smallproj<<<S_LEN, blk, 0, stream>>>(xn, lqa_w, lqa_b, tq, HID, 0);
  gemm(xn, HID, qkv_wt, HID, qkv, QKV_N, S_LEN, QKV_N, HID,
       qkv_b, tq, lqb_w, lqb_b, nullptr, 1.0f, 0);

  k_rope<<<dim3(S_LEN, QH + KVH), 64, 0, stream>>>(qkv, qr, kr);
  // V^T (bf16): vt[c][s] = qkv[s][2560 + c]
  k_w2bf16t<<<dim3(KVH * HD / 32, S_LEN / 32), blk, 0, stream>>>(
      qkv + (QH + KVH) * HD, vt, S_LEN, KVH * HD, QKV_N);
  k_smallproj<<<S_LEN, blk, 0, stream>>>(xn, gate_w, gate_b, gates, HID, 1);

  for (int h = 0; h < QH; ++h) {
    const unsigned short* qh = qr + h * HD;
    const unsigned short* kh = kr + (h / (QH / KVH)) * HD;
    const unsigned short* vh = vt + (long)(h / (QH / KVH)) * HD * S_LEN;
    gemm(qh, QH * HD, kh, KVH * HD, sc, S_LEN, S_LEN, S_LEN, HD,
         nullptr, nullptr, nullptr, nullptr, nullptr, SCALER, 1);
    k_softmax_bf16<<<S_LEN, blk, 0, stream>>>(sc, probs, S_LEN);
    gemm(probs, S_LEN, vh, S_LEN, av + h * HD, HID, S_LEN, HD, S_LEN,
         nullptr, nullptr, nullptr, nullptr, nullptr, 1.0f, 0);
  }

  k_gateapply<<<(S_LEN * HID) / 256, blk, 0, stream>>>(av, gates, av_bf);
  k_smallproj<<<S_LEN, blk, 0, stream>>>(av_bf, loa_w, loa_b, to_, HID, 0);
  gemm(av_bf, HID, o_wt, HID, hres, HID, S_LEN, HID, HID,
       o_b, to_, lob_w, lob_b, x, 1.0f, 0);          // hres = x + attn_out

  // ---- feed-forward branch ----
  k_rmsnorm_bf16<<<S_LEN, blk, 0, stream>>>(hres, n2w, xn2, HID);
  k_smallproj<<<S_LEN, blk, 0, stream>>>(xn2, lga_w, lga_b, tg, HID, 0);
  gemm(xn2, HID, gu_wt, HID, gu, 2 * FF, S_LEN, 2 * FF, HID,
       gu_b, tg, lgb_w, lgb_b, nullptr, 1.0f, 0);
  k_geluup<<<(S_LEN * FF) / 256, blk, 0, stream>>>(gu, hff);
  k_smallproj<<<S_LEN, blk, 0, stream>>>(hff, ldna_w, ldna_b, td, FF, 0);
  gemm(hff, FF, dn_wt, FF, out, HID, S_LEN, HID, FF,
       dn_b, td, ldnb_w, ldnb_b, hres, 1.0f, 0);     // out = hres + ffn_out
}
